// LstmSequential_30099130810549
// MI455X (gfx1250) — matrix operand order repaired and checked by reading, not compiled, not run on hardware
//
#include <hip/hip_runtime.h>
#include <hip/hip_bf16.h>

typedef _Float16 v16h __attribute__((ext_vector_type(16)));
typedef _Float16 v8h  __attribute__((ext_vector_type(8)));
typedef float    v8f  __attribute__((ext_vector_type(8)));

#define LSTM_SEQ    80
#define LSTM_EMB    100
#define LSTM_UNITS  64
#define NGATES      256           // 4*UNITS
#define K1TOT       192           // 128 (x padded 100->128) + 64 (h1)
#define K2TOT       128           // 64 (h1) + 64 (h2)
#define B1_FRAGS    (6 * 16)      // 6 k-chunks x 16 n-tiles
#define B2_FRAGS    (4 * 16)      // 4 k-chunks x 16 n-tiles
#define B1_HALVES   (B1_FRAGS * 32 * 16)   // 49152
#define B2_HALVES   (B2_FRAGS * 32 * 16)   // 32768

__device__ __forceinline__ float sigf(float x) {
  return 1.0f / (1.0f + __expf(-x));
}
__device__ __forceinline__ float tanhfast(float x) {
  x = fminf(fmaxf(x, -15.0f), 15.0f);
  float e = __expf(2.0f * x);
  return (e - 1.0f) / (e + 1.0f);
}

__device__ __forceinline__ v8f wmma_f16(v16h a, v16h b, v8f c) {
  // D = A(16x32 f16) * B(32x16 f16) + C(16x16 f32)
  return __builtin_amdgcn_wmma_f32_16x16x32_f16(false, a, false, b, (short)0, c,
                                                false, false);
}

// Build an A fragment (16x32 f16) from a row-major f16 LDS tile.
// Contiguous-k layout used here (same as packed B): lanes 0-15 hold row=lane, K = 0..15;
//             lanes 16-31 hold row=lane-16, K = 16..31.
__device__ __forceinline__ v16h load_afrag(const _Float16* s, int rowStride,
                                           int kc, int lane) {
  const int row = lane & 15;
  const int k0  = (lane >> 4) * 16;
  const _Float16* p = s + row * rowStride + kc * 32;
  v8h lo = *(const v8h*)(p + k0);
  v8h hi = *(const v8h*)(p + 8 + k0);
  v16h a;
#pragma unroll
  for (int j = 0; j < 8; ++j) { a[j] = lo[j]; a[8 + j] = hi[j]; }
  return a;
}

// -------- kernel 1: pack fused [W;U] weights into WMMA-B fragment-major f16 --------
// B fragment layout (32x16 f16): lane 0-15 = col (lane&15), K = kc*32 + 0..15;
//                                lane 16-31 = col, K = kc*32 + 16..31.
// Packed address: ((frag * 32 + lane) * 16 + j) halves, j = K offset 0..15.
__global__ void lstm_pack_weights(const float* __restrict__ W1,
                                  const float* __restrict__ U1,
                                  const float* __restrict__ W2,
                                  const float* __restrict__ U2,
                                  _Float16* __restrict__ B1p,
                                  _Float16* __restrict__ B2p) {
  int tid = blockIdx.x * 256 + threadIdx.x;  // one thread per (frag, lane)
  const int n1 = B1_FRAGS * 32;
  if (tid < n1) {
    int f = tid >> 5, lane = tid & 31;
    int kc = f >> 4, nt = f & 15;
    int col   = nt * 16 + (lane & 15);
    int kbase = kc * 32 + (lane >> 4) * 16;
#pragma unroll
    for (int j = 0; j < 16; ++j) {
      int k = kbase + j;
      float v;
      if (k < LSTM_EMB)      v = W1[k * NGATES + col];
      else if (k < 128)      v = 0.0f;                       // K pad 100->128
      else                   v = U1[(k - 128) * NGATES + col];
      B1p[(f * 32 + lane) * 16 + j] = (_Float16)v;
    }
  } else if (tid < n1 + B2_FRAGS * 32) {
    int t2 = tid - n1;
    int f = t2 >> 5, lane = t2 & 31;
    int kc = f >> 4, nt = f & 15;
    int col   = nt * 16 + (lane & 15);
    int kbase = kc * 32 + (lane >> 4) * 16;
#pragma unroll
    for (int j = 0; j < 16; ++j) {
      int k = kbase + j;
      float v = (k < 64) ? W2[k * NGATES + col] : U2[(k - 64) * NGATES + col];
      B2p[(f * 32 + lane) * 16 + j] = (_Float16)v;
    }
  }
}

// -------- kernel 2: fused 2-layer LSTM scan, 16 batch rows per workgroup --------
__global__ __launch_bounds__(256)
void lstm_scan_kernel(const int* __restrict__ tokens,
                      const float* __restrict__ emb,
                      const float* __restrict__ b1,
                      const float* __restrict__ b2,
                      const float* __restrict__ Wd,
                      const float* __restrict__ bd,
                      const _Float16* __restrict__ B1p,
                      const _Float16* __restrict__ B2p,
                      float* __restrict__ out) {
  __shared__ alignas(32) _Float16 sA1[16 * K1TOT];   // [x_pad(128) | h1(64)] per row
  __shared__ alignas(32) _Float16 sA2[16 * K2TOT];   // [h1(64) | h2(64)] per row
  __shared__ alignas(16) float    sZ[16 * NGATES];   // gate pre-activations
  __shared__ float sC1[16 * LSTM_UNITS];
  __shared__ float sC2[16 * LSTM_UNITS];

  const int tid  = threadIdx.x;
  const int lane = tid & 31;
  const int wave = tid >> 5;           // 8 waves, wave w owns n-tiles {2w, 2w+1}
  const int base = blockIdx.x * 16;    // batch rows [base, base+16)

  // zero-init: x pad region, h1/h2 = 0, c1/c2 = 0
  for (int i = tid; i < 16 * K1TOT; i += 256) sA1[i] = (_Float16)0.0f;
  for (int i = tid; i < 16 * K2TOT; i += 256) sA2[i] = (_Float16)0.0f;
  for (int i = tid; i < 16 * LSTM_UNITS; i += 256) { sC1[i] = 0.0f; sC2[i] = 0.0f; }

  // preload register-resident B fragments (reused for all 80 timesteps)
  v16h B1f[6][2];
  v16h B2f[4][2];
#pragma unroll
  for (int kc = 0; kc < 6; ++kc)
#pragma unroll
    for (int n = 0; n < 2; ++n) {
      int nt = wave * 2 + n;
      B1f[kc][n] = *(const v16h*)(B1p + ((kc * 16 + nt) * 32 + lane) * 16);
    }
#pragma unroll
  for (int kc = 0; kc < 4; ++kc)
#pragma unroll
    for (int n = 0; n < 2; ++n) {
      int nt = wave * 2 + n;
      B2f[kc][n] = *(const v16h*)(B2p + ((kc * 16 + nt) * 32 + lane) * 16);
    }

  // per-lane gate bias (C/D layout: lane -> column)
  float bias1[2], bias2[2];
#pragma unroll
  for (int n = 0; n < 2; ++n) {
    int col = (wave * 2 + n) * 16 + (lane & 15);
    bias1[n] = b1[col];
    bias2[n] = b2[col];
  }

  // prime the prefetcher for the first timestep's embedding rows
  {
    const int r = tid >> 4, c = tid & 15;
    if (c == 0) {
      int tok0 = tokens[(base + r) * LSTM_SEQ + 0];
      __builtin_prefetch(emb + (long)tok0 * LSTM_EMB, 0, 0);
    }
  }

  __syncthreads();

  const int zrow = (lane >> 4) * 8;             // D layout: VGPR r -> M = zrow + r
  const int zcol = wave * 32 + (lane & 15);     // n-tile pair base column

  for (int t = 0; t < LSTM_SEQ; ++t) {
    // ---- gather x_t = emb[tokens[:, t]] -> f16 into sA1[:, 0:100],
    //      and prefetch next step's embedding rows into L2/WGP$ ----
    {
      const int r = tid >> 4, c = tid & 15;
      const int tok = tokens[(base + r) * LSTM_SEQ + t];
      const float* er = emb + (long)tok * LSTM_EMB;
      for (int e = c; e < LSTM_EMB; e += 16)
        sA1[r * K1TOT + e] = (_Float16)er[e];
      if (c == 0 && t + 1 < LSTM_SEQ) {
        int tok2 = tokens[(base + r) * LSTM_SEQ + t + 1];
        __builtin_prefetch(emb + (long)tok2 * LSTM_EMB, 0, 0);
      }
    }
    __syncthreads();

    // ---- layer1 GEMM: z1 = [x|h1] @ [W1;U1] + b1  (K=192) ----
    // batch-issue all 12 ds_load_b128 first so the 12 WMMAs drain back-to-back
    {
      v16h a[6];
#pragma unroll
      for (int kc = 0; kc < 6; ++kc) a[kc] = load_afrag(sA1, K1TOT, kc, lane);
      v8f acc0, acc1;
#pragma unroll
      for (int r = 0; r < 8; ++r) { acc0[r] = bias1[0]; acc1[r] = bias1[1]; }
#pragma unroll
      for (int kc = 0; kc < 6; ++kc) {
        acc0 = wmma_f16(a[kc], B1f[kc][0], acc0);
        acc1 = wmma_f16(a[kc], B1f[kc][1], acc1);
      }
#pragma unroll
      for (int r = 0; r < 8; ++r) {
        sZ[(zrow + r) * NGATES + zcol]      = acc0[r];
        sZ[(zrow + r) * NGATES + zcol + 16] = acc1[r];
      }
    }
    __syncthreads();

    // ---- layer1 gates: c1,h1 update; h1 -> sA1 h-section + sA2[:,0:64] ----
#pragma unroll
    for (int k = 0; k < 4; ++k) {
      int e = tid + 256 * k;            // 1024 = 16 rows * 64 units
      int m = e >> 6, u = e & 63;
      const float* zr = sZ + m * NGATES;
      float zi = zr[u], zf = zr[64 + u], zg = zr[128 + u], zo = zr[192 + u];
      float c = sigf(zf) * sC1[e] + sigf(zi) * tanhfast(zg);
      float h = sigf(zo) * tanhfast(c);
      sC1[e] = c;
      _Float16 hh = (_Float16)h;
      sA1[m * K1TOT + 128 + u] = hh;
      sA2[m * K2TOT + u]       = hh;
    }
    __syncthreads();

    // ---- layer2 GEMM: z2 = [h1|h2] @ [W2;U2] + b2  (K=128) ----
    {
      v16h a[4];
#pragma unroll
      for (int kc = 0; kc < 4; ++kc) a[kc] = load_afrag(sA2, K2TOT, kc, lane);
      v8f acc0, acc1;
#pragma unroll
      for (int r = 0; r < 8; ++r) { acc0[r] = bias2[0]; acc1[r] = bias2[1]; }
#pragma unroll
      for (int kc = 0; kc < 4; ++kc) {
        acc0 = wmma_f16(a[kc], B2f[kc][0], acc0);
        acc1 = wmma_f16(a[kc], B2f[kc][1], acc1);
      }
#pragma unroll
      for (int r = 0; r < 8; ++r) {
        sZ[(zrow + r) * NGATES + zcol]      = acc0[r];
        sZ[(zrow + r) * NGATES + zcol + 16] = acc1[r];
      }
    }
    __syncthreads();

    // ---- layer2 gates: c2,h2 update; h2 -> sA2[:,64:128] ----
#pragma unroll
    for (int k = 0; k < 4; ++k) {
      int e = tid + 256 * k;
      int m = e >> 6, u = e & 63;
      const float* zr = sZ + m * NGATES;
      float zi = zr[u], zf = zr[64 + u], zg = zr[128 + u], zo = zr[192 + u];
      float c = sigf(zf) * sC2[e] + sigf(zi) * tanhfast(zg);
      float h = sigf(zo) * tanhfast(c);
      sC2[e] = c;
      sA2[m * K2TOT + 64 + u] = (_Float16)h;
    }
    __syncthreads();
  }

  // ---- dense head: out = sigmoid(h2_last @ Wd + bd) ----
  if (tid < 16) {
    float acc = bd[0];
    const _Float16* h2 = sA2 + tid * K2TOT + 64;
#pragma unroll
    for (int u = 0; u < LSTM_UNITS; ++u) acc += (float)h2[u] * Wd[u];
    out[base + tid] = sigf(acc);
  }
}

extern "C" void kernel_launch(void* const* d_in, const int* in_sizes, int n_in,
                              void* d_out, int out_size, void* d_ws, size_t ws_size,
                              hipStream_t stream) {
  const int*   tokens = (const int*)d_in[0];
  const float* emb    = (const float*)d_in[1];
  const float* W1     = (const float*)d_in[2];
  const float* U1     = (const float*)d_in[3];
  const float* b1     = (const float*)d_in[4];
  const float* W2     = (const float*)d_in[5];
  const float* U2     = (const float*)d_in[6];
  const float* b2     = (const float*)d_in[7];
  const float* Wd     = (const float*)d_in[8];
  const float* bd     = (const float*)d_in[9];

  _Float16* B1p = (_Float16*)d_ws;
  _Float16* B2p = B1p + B1_HALVES;

  // one thread per (fragment, lane): (96+64)*32 = 5120 threads
  lstm_pack_weights<<<20, 256, 0, stream>>>(W1, U1, W2, U2, B1p, B2p);

  const int batch = in_sizes[0] / LSTM_SEQ;  // 4096
  lstm_scan_kernel<<<batch / 16, 256, 0, stream>>>(tokens, emb, b1, b2, Wd, bd,
                                                   B1p, B2p, (float*)d_out);
}
